// NystromAttention_54451595379167
// MI455X (gfx1250) — matrix operand order repaired and checked by reading, not compiled, not run on hardware
//
#include <hip/hip_runtime.h>
#include <stdint.h>
#include <math.h>

#define BB 4
#define NN 4096
#define DD 512
#define HH 8
#define DH 64
#define MM 256
#define LL 16
#define INNER 512
#define KCONV 33
#define BH (BB*HH)

typedef __attribute__((ext_vector_type(16))) __bf16 v16bf;
typedef __attribute__((ext_vector_type(4)))  __bf16 bf16x4;
typedef __attribute__((ext_vector_type(8)))  float  v8f;
typedef __bf16 bf16;

// ---- WMMA fragment gathers (layouts per CDNA5 ISA 7.12.2, wave32) ----
__device__ __forceinline__ v16bf frag_a(const bf16* t, int ldk) {
  int lane = threadIdx.x & 31;
  int r = lane & 15, g = lane >> 4;
  v16bf f;
#pragma unroll
  for (int v = 0; v < 8; ++v) {
    int kb = ((v < 4) ? (2*v) : (16 + 2*(v-4))) + 8*g;
    f[2*v]   = t[r*ldk + kb];
    f[2*v+1] = t[r*ldk + kb + 1];
  }
  return f;
}
__device__ __forceinline__ v16bf frag_b(const bf16* t, int ldk) {
  int lane = threadIdx.x & 31;
  int c = lane & 15, g = lane >> 4;
  v16bf f;
#pragma unroll
  for (int v = 0; v < 8; ++v) {
    int kb = ((v < 4) ? (2*v) : (16 + 2*(v-4))) + 8*g;
    f[2*v]   = t[c*ldk + kb];
    f[2*v+1] = t[c*ldk + kb + 1];
  }
  return f;
}
__device__ __forceinline__ v8f wmma_bf16(v16bf a, v16bf b, v8f c) {
  return __builtin_amdgcn_wmma_f32_16x16x32_bf16(false, a, false, b, (short)0, c, false, false);
}

// ---- staging: load-phase batched into registers, then store-phase (single wait) ----
__device__ __forceinline__ void stage_a64_bf16(bf16* As, const bf16* src, size_t ld) {
  int tid = threadIdx.x;
  uint4 t[2];
#pragma unroll
  for (int it = 0; it < 2; ++it) {
    int idx = tid + it*128, r = idx >> 2, kc = (idx & 3) * 8;
    t[it] = *(const uint4*)(src + (size_t)r*ld + kc);
  }
#pragma unroll
  for (int it = 0; it < 2; ++it) {
    int idx = tid + it*128, r = idx >> 2, kc = (idx & 3) * 8;
    *(uint4*)(As + r*32 + kc) = t[it];
  }
}
__device__ __forceinline__ void stage_b_bf16(bf16* Bs, const bf16* src, size_t ld) {
  int tid = threadIdx.x;
  uint4 t[2];
#pragma unroll
  for (int it = 0; it < 2; ++it) {
    int idx = tid + it*128, kk = idx >> 3, cc = (idx & 7) * 8;
    t[it] = *(const uint4*)(src + (size_t)kk*ld + cc);
  }
#pragma unroll
  for (int it = 0; it < 2; ++it) {
    int idx = tid + it*128, kk = idx >> 3, cc = (idx & 7) * 8;
    const __bf16* p = (const __bf16*)&t[it];
#pragma unroll
    for (int j = 0; j < 8; ++j) Bs[(cc+j)*32 + kk] = p[j];
  }
}
__device__ __forceinline__ void stage_a64_f32(bf16* As, const float* src, int ld) {
  int tid = threadIdx.x;
  float4 t[4];
#pragma unroll
  for (int it = 0; it < 4; ++it) {
    int idx = tid + it*128, r = idx >> 3, kc = (idx & 7) * 4;
    t[it] = *(const float4*)(src + (size_t)r*ld + kc);
  }
#pragma unroll
  for (int it = 0; it < 4; ++it) {
    int idx = tid + it*128, r = idx >> 3, kc = (idx & 7) * 4;
    bf16x4 o; o[0]=(__bf16)t[it].x; o[1]=(__bf16)t[it].y; o[2]=(__bf16)t[it].z; o[3]=(__bf16)t[it].w;
    *(bf16x4*)(As + r*32 + kc) = o;
  }
}
__device__ __forceinline__ void stage_b_f32(bf16* Bs, const float* src, int ld) {
  int tid = threadIdx.x;
  float4 t[4];
#pragma unroll
  for (int it = 0; it < 4; ++it) {
    int idx = tid + it*128, kk = idx >> 4, cc = (idx & 15) * 4;
    t[it] = *(const float4*)(src + (size_t)kk*ld + cc);
  }
#pragma unroll
  for (int it = 0; it < 4; ++it) {
    int idx = tid + it*128, kk = idx >> 4, cc = (idx & 15) * 4;
    Bs[(cc+0)*32 + kk] = (__bf16)t[it].x;
    Bs[(cc+1)*32 + kk] = (__bf16)t[it].y;
    Bs[(cc+2)*32 + kk] = (__bf16)t[it].z;
    Bs[(cc+3)*32 + kk] = (__bf16)t[it].w;
  }
}
// 64x64 block MMA: wave owns col tile, 4 row sub-tiles; B frag reused 4x
__device__ __forceinline__ void mma_64x64(const bf16* As, const bf16* Bs, v8f acc[4], int wave) {
  v16bf fb = frag_b(Bs + wave*16*32, 32);
#pragma unroll
  for (int r = 0; r < 4; ++r)
    acc[r] = wmma_bf16(frag_a(As + r*16*32, 32), fb, acc[r]);
}
// double-buffered K loops: stage tile s+1 while computing tile s (1 barrier/step)
__device__ __forceinline__ void gemm_loop_bf16(const bf16* Asrc, size_t lda,
                                               const bf16* Bsrc, size_t ldb,
                                               int ksteps, v8f acc[4], int wave,
                                               bf16 As[2][64*32], bf16 Bs[2][64*32]) {
  stage_a64_bf16(As[0], Asrc, lda);
  stage_b_bf16(Bs[0], Bsrc, ldb);
  __syncthreads();
  for (int s = 0; s < ksteps; ++s) {
    int cur = s & 1;
    if (s + 1 < ksteps) {
      stage_a64_bf16(As[cur^1], Asrc + (size_t)(s+1)*32, lda);
      stage_b_bf16(Bs[cur^1], Bsrc + (size_t)(s+1)*32*ldb, ldb);
    }
    if (s + 2 < ksteps) {
      int t = threadIdx.x;
      __builtin_prefetch(Asrc + (size_t)(t & 63)*lda + (s+2)*32, 0, 1);
      __builtin_prefetch(Bsrc + (size_t)((s+2)*32 + (t >> 2))*ldb, 0, 1);
    }
    mma_64x64(As[cur], Bs[cur], acc, wave);
    __syncthreads();
  }
}
__device__ __forceinline__ void gemm_loop_f32(const float* Asrc, int lda,
                                              const float* Bsrc, int ldb,
                                              int ksteps, v8f acc[4], int wave,
                                              bf16 As[2][64*32], bf16 Bs[2][64*32]) {
  stage_a64_f32(As[0], Asrc, lda);
  stage_b_f32(Bs[0], Bsrc, ldb);
  __syncthreads();
  for (int s = 0; s < ksteps; ++s) {
    int cur = s & 1;
    if (s + 1 < ksteps) {
      stage_a64_f32(As[cur^1], Asrc + (s+1)*32, lda);
      stage_b_f32(Bs[cur^1], Bsrc + (size_t)(s+1)*32*ldb, ldb);
    }
    if (s + 2 < ksteps) {
      int t = threadIdx.x;
      __builtin_prefetch(Asrc + (size_t)(t & 63)*lda + (s+2)*32, 0, 1);
      __builtin_prefetch(Bsrc + (size_t)((s+2)*32 + (t >> 2))*ldb, 0, 1);
    }
    mma_64x64(As[cur], Bs[cur], acc, wave);
    __syncthreads();
  }
}

// ---- 0. vectorized f32 -> bf16 conversion (one-time per tensor) ----
__global__ void k_cvt(const float* __restrict__ s, bf16* __restrict__ d, int n4) {
  int i = blockIdx.x*blockDim.x + threadIdx.x;
  if (i >= n4) return;
  float4 v = ((const float4*)s)[i];
  bf16x4 o; o[0]=(__bf16)v.x; o[1]=(__bf16)v.y; o[2]=(__bf16)v.z; o[3]=(__bf16)v.w;
  ((bf16x4*)d)[i] = o;
}

// ---- 1. qkv = x @ W_qkv, scatter into q/k/v [BH][N][DH] bf16 (q * DH^-0.5) ----
__global__ __launch_bounds__(128) void k_qkv(const bf16* __restrict__ xb,
                                             const bf16* __restrict__ wqkvb,
                                             bf16* __restrict__ qb, bf16* __restrict__ kb,
                                             bf16* __restrict__ vb) {
  __shared__ __align__(16) bf16 As[2][64*32];
  __shared__ __align__(16) bf16 Bs[2][64*32];
  int tid = threadIdx.x, wave = tid >> 5, lane = tid & 31;
  int row0 = blockIdx.y * 64, col0 = blockIdx.x * 64;
  v8f acc[4] = {};
  gemm_loop_bf16(xb + (size_t)row0*DD, DD, wqkvb + col0, 3*INNER, DD/32, acc, wave, As, Bs);
  int g = lane >> 4, c15 = lane & 15;
  int c = col0 + wave*16 + c15;
  int part = c / INNER, rem = c % INNER;
  int h = rem / DH, dh = rem % DH;
#pragma unroll
  for (int r = 0; r < 4; ++r) {
#pragma unroll
    for (int e = 0; e < 8; ++e) {
      int rw = row0 + r*16 + e + 8*g;
      int b = rw / NN, n = rw % NN;
      size_t dst = (((size_t)(b*HH + h))*NN + n)*DH + dh;
      float val = acc[r][e];
      if (part == 0)      qb[dst] = (bf16)(val * 0.125f);
      else if (part == 1) kb[dst] = (bf16)val;
      else                vb[dst] = (bf16)val;
    }
  }
}

// ---- 2. landmarks: mean over 16 tokens ----
__global__ void k_landmark(const bf16* __restrict__ qb, const bf16* __restrict__ kb,
                           bf16* __restrict__ qlb, bf16* __restrict__ klb) {
  int idx = blockIdx.x*blockDim.x + threadIdx.x;
  if (idx >= BH*MM*DH) return;
  int d = idx % DH, m = (idx / DH) % MM, bh = idx / (DH*MM);
  size_t base = ((size_t)bh*NN + (size_t)m*LL)*DH + d;
  float sq = 0.f, sk = 0.f;
  for (int j = 0; j < LL; ++j) { sq += (float)qb[base + (size_t)j*DH]; sk += (float)kb[base + (size_t)j*DH]; }
  size_t o = ((size_t)bh*MM + m)*DH + d;
  qlb[o] = (bf16)(sq * (1.0f/LL));
  klb[o] = (bf16)(sk * (1.0f/LL));
}

// ---- 3. softmax(Q @ KL^T) over 256-wide rows (sim1 -> attn1 bf16, sim2 -> x2 f32) ----
__global__ __launch_bounds__(128) void k_attn_rows256(const bf16* __restrict__ Q, int nrows,
                                                      const bf16* __restrict__ KL,
                                                      bf16* __restrict__ out_bf,
                                                      float* __restrict__ out_f32) {
  __shared__ __align__(16) bf16 Qs[16*64];
  __shared__ __align__(16) bf16 Ks[256*64];
  __shared__ float S[16*256];
  __shared__ float red[128];
  __shared__ float mrow[16], srow[16];
  int tid = threadIdx.x, wave = tid >> 5, lane = tid & 31;
  int bh = blockIdx.y, row0 = blockIdx.x * 16;
  const bf16* Qbase = Q + ((size_t)bh*nrows + row0)*DH;
  const bf16* Kbase = KL + (size_t)bh*MM*DH;
  ((uint4*)Qs)[tid & 127] = ((const uint4*)Qbase)[tid & 127];
  for (int i = tid; i < 2048; i += 128) ((uint4*)Ks)[i] = ((const uint4*)Kbase)[i];
  __syncthreads();
  int g = lane >> 4, c15 = lane & 15;
  for (int ct = 0; ct < 4; ++ct) {
    int col0 = wave*64 + ct*16;
    v8f acc = {};
    acc = wmma_bf16(frag_a(Qs, 64),      frag_b(Ks + col0*64, 64),      acc);
    acc = wmma_bf16(frag_a(Qs + 32, 64), frag_b(Ks + col0*64 + 32, 64), acc);
#pragma unroll
    for (int e = 0; e < 8; ++e) S[(e + 8*g)*256 + col0 + c15] = acc[e];
  }
  __syncthreads();
  // parallel row softmax: 8 segments of 32 per row
  int rr = tid >> 3, sg = tid & 7;
  float mloc = -1e30f;
  for (int c = sg*32; c < sg*32 + 32; ++c) mloc = fmaxf(mloc, S[rr*256 + c]);
  red[tid] = mloc;
  __syncthreads();
  if (tid < 16) {
    float mm = red[tid*8];
    for (int j = 1; j < 8; ++j) mm = fmaxf(mm, red[tid*8 + j]);
    mrow[tid] = mm;
  }
  __syncthreads();
  float mr = mrow[rr], sloc = 0.f;
  for (int c = sg*32; c < sg*32 + 32; ++c) sloc += expf(S[rr*256 + c] - mr);
  red[tid] = sloc;
  __syncthreads();
  if (tid < 16) {
    float ss = 0.f;
    for (int j = 0; j < 8; ++j) ss += red[tid*8 + j];
    srow[tid] = 1.0f / ss;
  }
  __syncthreads();
  float inv = srow[rr];
  for (int c = sg*32; c < sg*32 + 32; ++c) {
    float pv = expf(S[rr*256 + c] - mr) * inv;
    size_t o = ((size_t)bh*nrows + row0 + rr)*MM + c;
    if (out_bf) out_bf[o] = (bf16)pv; else out_f32[o] = pv;
  }
}

// ---- 4. softmax(QL @ K^T) over 4096-wide rows, online stats + recompute (sim3 -> attn3) ----
__global__ __launch_bounds__(128) void k_attn_rows4096(const bf16* __restrict__ QL,
                                                       const bf16* __restrict__ Kf,
                                                       bf16* __restrict__ out_bf) {
  __shared__ __align__(16) bf16 Qs[16*64];
  __shared__ __align__(16) bf16 Ks[256*64];
  __shared__ float S[16*256];
  __shared__ float red[128];
  __shared__ float mrow[16], srow[16], tmn[16];
  int tid = threadIdx.x, wave = tid >> 5, lane = tid & 31;
  int bh = blockIdx.y, row0 = blockIdx.x * 16;
  const bf16* Qbase = QL + ((size_t)bh*MM + row0)*DH;
  ((uint4*)Qs)[tid & 127] = ((const uint4*)Qbase)[tid & 127];
  if (tid < 16) { mrow[tid] = -1e30f; srow[tid] = 0.f; }
  __syncthreads();
  int g = lane >> 4, c15 = lane & 15;
  int rr = tid >> 3, sg = tid & 7;
  for (int pass = 0; pass < 2; ++pass) {
    for (int ch = 0; ch < NN/256; ++ch) {
      const bf16* Kbase = Kf + ((size_t)bh*NN + (size_t)ch*256)*DH;
      for (int i = tid; i < 2048; i += 128) ((uint4*)Ks)[i] = ((const uint4*)Kbase)[i];
      __syncthreads();
      for (int ct = 0; ct < 4; ++ct) {
        int col0 = wave*64 + ct*16;
        v8f acc = {};
        acc = wmma_bf16(frag_a(Qs, 64),      frag_b(Ks + col0*64, 64),      acc);
        acc = wmma_bf16(frag_a(Qs + 32, 64), frag_b(Ks + col0*64 + 32, 64), acc);
#pragma unroll
        for (int e = 0; e < 8; ++e) S[(e + 8*g)*256 + col0 + c15] = acc[e];
      }
      __syncthreads();
      if (pass == 0) {                       // online max/sum update (parallel)
        float mloc = -1e30f;
        for (int c = sg*32; c < sg*32 + 32; ++c) mloc = fmaxf(mloc, S[rr*256 + c]);
        red[tid] = mloc;
        __syncthreads();
        if (tid < 16) {
          float cm = red[tid*8];
          for (int j = 1; j < 8; ++j) cm = fmaxf(cm, red[tid*8 + j]);
          tmn[tid] = fmaxf(mrow[tid], cm);
        }
        __syncthreads();
        float mn = tmn[rr], sloc = 0.f;
        for (int c = sg*32; c < sg*32 + 32; ++c) sloc += expf(S[rr*256 + c] - mn);
        red[tid] = sloc;
        __syncthreads();
        if (tid < 16) {
          float tot = 0.f;
          for (int j = 0; j < 8; ++j) tot += red[tid*8 + j];
          srow[tid] = srow[tid] * expf(mrow[tid] - tmn[tid]) + tot;
          mrow[tid] = tmn[tid];
        }
      } else {                               // normalized write
        float mr = mrow[rr], inv = 1.0f / srow[rr];
        for (int c = sg*32; c < sg*32 + 32; ++c) {
          float pv = expf(S[rr*256 + c] - mr) * inv;
          out_bf[((size_t)bh*MM + row0 + rr)*NN + (size_t)ch*256 + c] = (bf16)pv;
        }
      }
      __syncthreads();
    }
  }
}

// ---- 5. kv = attn3 @ v : [256 x 64] per (b,h), K = 4096 ----
__global__ __launch_bounds__(128) void k_gemm_kv(const bf16* __restrict__ attn3,
                                                 const bf16* __restrict__ vb,
                                                 bf16* __restrict__ kvb) {
  __shared__ __align__(16) bf16 As[2][64*32];
  __shared__ __align__(16) bf16 Bs[2][64*32];
  int tid = threadIdx.x, wave = tid >> 5, lane = tid & 31;
  int bh = blockIdx.y, row0 = blockIdx.x * 64;
  v8f acc[4] = {};
  gemm_loop_bf16(attn3 + ((size_t)bh*MM + row0)*NN, NN,
                 vb + (size_t)bh*NN*DH, DH, NN/32, acc, wave, As, Bs);
  int g = lane >> 4, c15 = lane & 15;
#pragma unroll
  for (int r = 0; r < 4; ++r)
#pragma unroll
    for (int e = 0; e < 8; ++e)
      kvb[((size_t)bh*MM + row0 + r*16 + e + 8*g)*DH + wave*16 + c15] = (bf16)acc[r][e];
}

// ---- 6. pinv: global norm + init + Newton-Schulz steps ----
__global__ void k_norm1(const float* __restrict__ x, float* __restrict__ partial) {
  __shared__ float rs[256], cs[256];
  int bh = blockIdx.x, t = threadIdx.x;
  const float* Xm = x + (size_t)bh*MM*MM;
  float rsum = 0.f, csum = 0.f;
  for (int j = 0; j < MM; ++j) { rsum += Xm[t*MM + j]; csum += Xm[j*MM + t]; }
  rs[t] = rsum; cs[t] = csum;
  __syncthreads();
  for (int s = 128; s > 0; s >>= 1) {
    if (t < s) { rs[t] = fmaxf(rs[t], rs[t+s]); cs[t] = fmaxf(cs[t], cs[t+s]); }
    __syncthreads();
  }
  if (t == 0) { partial[bh] = rs[0]; partial[BH + bh] = cs[0]; }
}
__global__ void k_norm2(const float* __restrict__ partial, float* __restrict__ norm) {
  if (threadIdx.x == 0) {
    float mr = -1e30f, mc = -1e30f;
    for (int i = 0; i < BH; ++i) { mr = fmaxf(mr, partial[i]); mc = fmaxf(mc, partial[BH + i]); }
    norm[0] = 1.0f / (mr * mc);
  }
}
__global__ void k_pinv_init(const float* __restrict__ x, const float* __restrict__ norm,
                            float* __restrict__ z) {
  int idx = blockIdx.x*blockDim.x + threadIdx.x;
  if (idx >= BH*MM*MM) return;
  int j = idx % MM, i = (idx / MM) % MM, bh = idx / (MM*MM);
  z[((size_t)bh*MM + i)*MM + j] = x[((size_t)bh*MM + j)*MM + i] * norm[0];
}
// C = scale * (alpha*A + beta*(A@B)); identity-shift fused in f32 epilogue
__global__ __launch_bounds__(128) void k_pinv_gemm(const float* __restrict__ A,
                                                   const float* __restrict__ Bm,
                                                   float* __restrict__ C,
                                                   float alpha, float beta, float scale) {
  __shared__ __align__(16) bf16 As[2][64*32];
  __shared__ __align__(16) bf16 Bs[2][64*32];
  int tid = threadIdx.x, wave = tid >> 5, lane = tid & 31;
  int bh = blockIdx.y;
  int row0 = (blockIdx.x >> 2) * 64;
  int col0 = (blockIdx.x & 3) * 64;
  const float* Am = A + (size_t)bh*MM*MM;
  const float* Bb = Bm + (size_t)bh*MM*MM;
  v8f acc[4] = {};
  gemm_loop_f32(Am + (size_t)row0*MM, MM, Bb + col0, MM, MM/32, acc, wave, As, Bs);
  int g = lane >> 4, c15 = lane & 15;
  int col = col0 + wave*16 + c15;
#pragma unroll
  for (int r = 0; r < 4; ++r) {
#pragma unroll
    for (int e = 0; e < 8; ++e) {
      int row = row0 + r*16 + e + 8*g;
      float a = (alpha != 0.f) ? Am[(size_t)row*MM + col] : 0.f;
      C[(size_t)bh*MM*MM + (size_t)row*MM + col] = scale * (alpha * a + beta * acc[r][e]);
    }
  }
}

// ---- 7. out1 = attn1 @ z_bf -> bf16 [BH][N][M] ----
__global__ __launch_bounds__(128) void k_gemm_out1(const bf16* __restrict__ attn1,
                                                   const bf16* __restrict__ zb,
                                                   bf16* __restrict__ out1) {
  __shared__ __align__(16) bf16 As[2][64*32];
  __shared__ __align__(16) bf16 Bs[2][64*32];
  int tid = threadIdx.x, wave = tid >> 5, lane = tid & 31;
  int bh = blockIdx.y;
  int row0 = (blockIdx.x >> 2) * 64;
  int col0 = (blockIdx.x & 3) * 64;
  v8f acc[4] = {};
  gemm_loop_bf16(attn1 + (size_t)bh*NN*MM + (size_t)row0*MM, MM,
                 zb + (size_t)bh*MM*MM + col0, MM, MM/32, acc, wave, As, Bs);
  int g = lane >> 4, c15 = lane & 15;
#pragma unroll
  for (int r = 0; r < 4; ++r)
#pragma unroll
    for (int e = 0; e < 8; ++e)
      out1[(size_t)bh*NN*MM + (size_t)(row0 + r*16 + e + 8*g)*MM + col0 + wave*16 + c15] = (bf16)acc[r][e];
}

// ---- 8. out = out1 @ kv + depthwise_conv(v) via LDS halo tile -> hid [B][N][H*DH] ----
__global__ __launch_bounds__(128) void k_gemm_out2_conv(const bf16* __restrict__ out1,
                                                        const bf16* __restrict__ kvb,
                                                        const bf16* __restrict__ vb,
                                                        const float* __restrict__ convw,
                                                        bf16* __restrict__ hid) {
  __shared__ __align__(16) bf16 As[2][64*32];
  __shared__ __align__(16) bf16 Bs[2][64*32];
  __shared__ __align__(16) bf16 Vs[96*64];
  int tid = threadIdx.x, wave = tid >> 5, lane = tid & 31;
  int bh = blockIdx.y;
  int h = bh % HH, b = bh / HH;
  int row0 = blockIdx.x * 64;
  const bf16* Vb = vb + (size_t)bh*NN*DH;
  for (int i = tid; i < 768; i += 128) {
    int rr = i >> 3, cc = (i & 7) * 8;
    int n = row0 - 16 + rr;
    uint4 u = {0u, 0u, 0u, 0u};
    if (n >= 0 && n < NN) u = *(const uint4*)(Vb + (size_t)n*DH + cc);
    *(uint4*)(Vs + rr*64 + cc) = u;
  }
  v8f acc[4] = {};
  gemm_loop_bf16(out1 + (size_t)bh*NN*MM + (size_t)row0*MM, MM,
                 kvb + (size_t)bh*MM*DH, DH, MM/32, acc, wave, As, Bs);
  int g = lane >> 4, c15 = lane & 15;
  int dh = wave*16 + c15;
#pragma unroll
  for (int r = 0; r < 4; ++r) {
#pragma unroll
    for (int e = 0; e < 8; ++e) {
      int ln = r*16 + e + 8*g;
      int n = row0 + ln;
      float acc_c = acc[r][e];
      for (int t = 0; t < KCONV; ++t)
        acc_c += convw[h*KCONV + t] * (float)Vs[(ln + t)*64 + dh];
      hid[((size_t)b*NN + n)*INNER + h*DH + dh] = (bf16)acc_c;
    }
  }
}

// ---- 9. final: out = hid @ W_out + b_out (f32 out) ----
__global__ __launch_bounds__(128) void k_gemm_final(const bf16* __restrict__ hid,
                                                    const bf16* __restrict__ woutb,
                                                    const float* __restrict__ bout,
                                                    float* __restrict__ out) {
  __shared__ __align__(16) bf16 As[2][64*32];
  __shared__ __align__(16) bf16 Bs[2][64*32];
  int tid = threadIdx.x, wave = tid >> 5, lane = tid & 31;
  int row0 = blockIdx.y * 64, col0 = blockIdx.x * 64;
  v8f acc[4] = {};
  gemm_loop_bf16(hid + (size_t)row0*INNER, INNER, woutb + col0, DD, INNER/32, acc, wave, As, Bs);
  int g = lane >> 4, c15 = lane & 15;
  int c = col0 + wave*16 + c15;
  float bo = bout[c];
#pragma unroll
  for (int r = 0; r < 4; ++r)
#pragma unroll
    for (int e = 0; e < 8; ++e)
      out[(size_t)(row0 + r*16 + e + 8*g)*DD + c] = acc[r][e] + bo;
}

extern "C" void kernel_launch(void* const* d_in, const int* in_sizes, int n_in,
                              void* d_out, int out_size, void* d_ws, size_t ws_size,
                              hipStream_t stream) {
  (void)in_sizes; (void)n_in; (void)out_size; (void)ws_size;
  const float* x     = (const float*)d_in[0];
  const float* wqkv  = (const float*)d_in[1];
  const float* wout  = (const float*)d_in[2];
  const float* bout  = (const float*)d_in[3];
  const float* convw = (const float*)d_in[4];
  float* out = (float*)d_out;

  char* ws = (char*)d_ws;
  size_t off = 0;
  auto alloc = [&](size_t bytes) -> char* {
    char* p = ws + off;
    off += (bytes + 255) & ~(size_t)255;
    return p;
  };
  bf16*  xb    = (bf16*) alloc((size_t)BB*NN*DD*sizeof(bf16));
  bf16*  wqkvb = (bf16*) alloc((size_t)DD*3*INNER*sizeof(bf16));
  bf16*  woutb = (bf16*) alloc((size_t)INNER*DD*sizeof(bf16));
  bf16*  qb    = (bf16*) alloc((size_t)BH*NN*DH*sizeof(bf16));
  bf16*  kb    = (bf16*) alloc((size_t)BH*NN*DH*sizeof(bf16));
  bf16*  vb    = (bf16*) alloc((size_t)BH*NN*DH*sizeof(bf16));
  bf16*  qlb   = (bf16*) alloc((size_t)BH*MM*DH*sizeof(bf16));
  bf16*  klb   = (bf16*) alloc((size_t)BH*MM*DH*sizeof(bf16));
  bf16*  attn1 = (bf16*) alloc((size_t)BH*NN*MM*sizeof(bf16));
  bf16*  attn3 = (bf16*) alloc((size_t)BH*NN*MM*sizeof(bf16));  // later reused as out1
  float* x2    = (float*)alloc((size_t)BH*MM*MM*sizeof(float));
  float* zA    = (float*)alloc((size_t)BH*MM*MM*sizeof(float));
  float* zB    = (float*)alloc((size_t)BH*MM*MM*sizeof(float));
  float* xz    = (float*)alloc((size_t)BH*MM*MM*sizeof(float));
  float* t1    = (float*)alloc((size_t)BH*MM*MM*sizeof(float));
  float* t2    = (float*)alloc((size_t)BH*MM*MM*sizeof(float));
  bf16*  zbf   = (bf16*) alloc((size_t)BH*MM*MM*sizeof(bf16));
  bf16*  kvb   = (bf16*) alloc((size_t)BH*MM*DH*sizeof(bf16));
  bf16*  hid   = (bf16*) alloc((size_t)BB*NN*INNER*sizeof(bf16));
  float* part  = (float*)alloc(2*BH*sizeof(float));
  float* norm  = (float*)alloc(256);

  dim3 blk(128);
  // one-time operand downconversions (TF32-style bf16 tensor-core path)
  k_cvt<<<dim3((BB*NN*DD/4 + 255)/256), dim3(256), 0, stream>>>(x, xb, BB*NN*DD/4);
  k_cvt<<<dim3((DD*3*INNER/4 + 255)/256), dim3(256), 0, stream>>>(wqkv, wqkvb, DD*3*INNER/4);
  k_cvt<<<dim3((INNER*DD/4 + 255)/256), dim3(256), 0, stream>>>(wout, woutb, INNER*DD/4);

  k_qkv<<<dim3(3*INNER/64, (BB*NN)/64), blk, 0, stream>>>(xb, wqkvb, qb, kb, vb);
  k_landmark<<<dim3((BH*MM*DH)/256), dim3(256), 0, stream>>>(qb, kb, qlb, klb);
  k_attn_rows256<<<dim3(NN/16, BH), blk, 0, stream>>>(qb, NN, klb, attn1, nullptr);
  k_attn_rows256<<<dim3(MM/16, BH), blk, 0, stream>>>(qlb, MM, klb, nullptr, x2);
  k_attn_rows4096<<<dim3(MM/16, BH), blk, 0, stream>>>(qlb, kb, attn3);
  k_gemm_kv<<<dim3(MM/64, BH), blk, 0, stream>>>(attn3, vb, kvb);

  k_norm1<<<dim3(BH), dim3(256), 0, stream>>>(x2, part);
  k_norm2<<<dim3(1), dim3(32), 0, stream>>>(part, norm);
  k_pinv_init<<<dim3((BH*MM*MM)/256), dim3(256), 0, stream>>>(x2, norm, zA);
  float* zc = zA; float* zn = zB;
  for (int it = 0; it < 6; ++it) {
    k_pinv_gemm<<<dim3(16, BH), blk, 0, stream>>>(x2, zc, xz, 0.f,  1.f, 1.f);   // xz = x@z
    k_pinv_gemm<<<dim3(16, BH), blk, 0, stream>>>(xz, xz, t1, 7.f,  -1.f, 1.f);  // 7xz - xz@xz
    k_pinv_gemm<<<dim3(16, BH), blk, 0, stream>>>(xz, t1, t2, 15.f, -1.f, 1.f);  // 15xz - xz@t1
    k_pinv_gemm<<<dim3(16, BH), blk, 0, stream>>>(zc, t2, zn, 13.f, -1.f, 0.25f);// .25(13z - z@t2)
    float* tmp = zc; zc = zn; zn = tmp;
  }
  k_cvt<<<dim3((BH*MM*MM/4 + 255)/256), dim3(256), 0, stream>>>(zc, zbf, BH*MM*MM/4);

  k_gemm_out1<<<dim3((NN/64)*4, BH), blk, 0, stream>>>(attn1, zbf, attn3);  // attn3 := out1
  k_gemm_out2_conv<<<dim3(NN/64, BH), blk, 0, stream>>>(attn3, kvb, vb, convw, hid);
  k_gemm_final<<<dim3(DD/64, (BB*NN)/64), blk, 0, stream>>>(hid, woutb, bout, out);
}
